// I2BGNN_27977416966480
// MI455X (gfx1250) — matrix operand order repaired and checked by reading, not compiled, not run on hardware
//
#include <hip/hip_runtime.h>
#include <hip/hip_bf16.h>

typedef __attribute__((ext_vector_type(16))) _Float16 v16h;
typedef __attribute__((ext_vector_type(8)))  _Float16 v8h;
typedef __attribute__((ext_vector_type(8)))  float    v8f;

#define DD 128   // feature dim (fixed by problem)
#define MT 4     // row tiles (of 16) per block in the WMMA GEMM

// ---------------------------------------------------------------- utilities
__global__ void zero_f32(float* __restrict__ p, long long n) {
    long long i = (long long)blockIdx.x * blockDim.x + threadIdx.x;
    if (i < n) p[i] = 0.0f;
}

__global__ void cvt_f16(const float* __restrict__ in, _Float16* __restrict__ out, long long n) {
    long long i = (long long)blockIdx.x * blockDim.x + threadIdx.x;
    if (i < n) out[i] = (_Float16)in[i];
}

// f32 row-major [rows x cols] -> f16 col-major [cols x rows] (weight transpose)
__global__ void cvt_f16_t(const float* __restrict__ in, _Float16* __restrict__ out,
                          int rows, int cols) {
    int i = blockIdx.x * blockDim.x + threadIdx.x;
    if (i >= rows * cols) return;
    int r = i / cols, c = i - r * cols;
    out[(size_t)c * rows + r] = (_Float16)in[i];
}

__global__ void fill_f32(float* __restrict__ p, float v, long long n) {
    long long i = (long long)blockIdx.x * blockDim.x + threadIdx.x;
    if (i < n) p[i] = v;
}

// ---------------------------------------------------------------- degree / norm
__global__ void deg_acc(const int* __restrict__ dst, const float* __restrict__ eattr,
                        float* __restrict__ deg, int E) {
    int e = blockIdx.x * blockDim.x + threadIdx.x;
    if (e < E) atomicAdd(&deg[dst[e]], eattr[2 * e + 1]);
}

__global__ void rsqrt_k(const float* __restrict__ deg, float* __restrict__ dinv, int n) {
    int i = blockIdx.x * blockDim.x + threadIdx.x;
    if (i < n) dinv[i] = __frsqrt_rn(deg[i]);
}

// ---------------------------------------------------------------- WMMA GEMM
// Out[M x 128] = A[M x 128](f16 row-major) @ W (passed as Wt = W^T, f16
// col-major), f32 accumulate via v_wmma_f32_16x16x32_f16.
//
// Block = 256 threads = 8 wave32 waves; wave w owns a 16-column tile; block
// covers MT*16 rows. Wt staged once in LDS (32 KB of 320 KB/WGP); K-contiguous
// columns make each B fragment one aligned 32-byte LDS vector. B fragments
// are held in VGPRs across all MT row tiles. Epilogue uses a uniform
// tile-level bound check so the hot path is 8 unguarded coalesced stores.
template <int DO_RELU, int HAS_BIAS>
__global__ void gemm_wmma_128(const _Float16* __restrict__ A,
                              const _Float16* __restrict__ Wt,
                              float* __restrict__ Out,
                              const float* __restrict__ bias,
                              int M)
{
    __shared__ _Float16 wl[DD * DD];
    {   // linear cooperative copy (Wt already fragment-ordered)
        const float4* s = (const float4*)Wt;
        float4*       d = (float4*)wl;
        for (int i = threadIdx.x; i < DD * DD * 2 / 16; i += blockDim.x)
            d[i] = s[i];
    }
    __syncthreads();

    const int wave = threadIdx.x >> 5;        // 0..7 -> column tile
    const int lane = threadIdx.x & 31;
    const int l16  = lane & 15;
    const int half = lane >> 4;               // lanes 0-15 vs 16-31
    const int col  = wave * 16 + l16;

    // B fragments: b[i] <-> k_local = half*16 + i (contiguous in col-major Wt)
    v16h bfrag[DD / 32];
    #pragma unroll
    for (int ks = 0; ks < DD / 32; ++ks)
        bfrag[ks] = *(const v16h*)(wl + (size_t)col * DD + ks * 32 + half * 16);

    const float bb = HAS_BIAS ? bias[col] : 0.0f;

    #pragma unroll
    for (int t = 0; t < MT; ++t) {
        const int row0 = (blockIdx.x * MT + t) * 16;
        if (row0 >= M) break;

        const int arow = min(row0 + l16, M - 1);
        const _Float16* ap = A + (size_t)arow * DD;

        v8f acc = {};
        #pragma unroll
        for (int ks = 0; ks < DD / 32; ++ks) {
            const int kk = ks * 32;
            // A fragment = two contiguous 16 B chunks per lane:
            //   a[0..7]  <-> K = kk + half*16 + {0..7}
            //   a[8..15] <-> K = kk + half*16 + 8 + {0..7}
            const v8h a0 = *(const v8h*)(ap + kk + half * 16);
            const v8h a1 = *(const v8h*)(ap + kk + half * 16 + 8);
            v16h a;
            #pragma unroll
            for (int i = 0; i < 8; ++i) { a[i] = a0[i]; a[8 + i] = a1[i]; }

            acc = __builtin_amdgcn_wmma_f32_16x16x32_f16(
                      false, a, false, bfrag[ks], (short)0, acc, false, false);
        }

        // C/D: lane -> tile col = l16, VGPR r -> row = r + 8*half
        float* outp = Out + (size_t)(row0 + 8 * half) * DD + col;
        if (row0 + 16 <= M) {               // uniform fast path: no per-row guard
            #pragma unroll
            for (int r = 0; r < 8; ++r) {
                float v = acc[r] + bb;
                if (DO_RELU) v = fmaxf(v, 0.0f);
                outp[(size_t)r * DD] = v;
            }
        } else {                            // ragged tail tile (rare/never here)
            #pragma unroll
            for (int r = 0; r < 8; ++r) {
                if (row0 + r + 8 * half < M) {
                    float v = acc[r] + bb;
                    if (DO_RELU) v = fmaxf(v, 0.0f);
                    outp[(size_t)r * DD] = v;
                }
            }
        }
    }
}

// ---------------------------------------------------------------- edge scatter
// 32 threads per edge, 4 floats each. Gather h[src] (51.2 MB of node features
// is L2-resident on the 192 MB L2), atomic scatter into agg[dst].
__global__ void edge_agg(const int* __restrict__ src, const int* __restrict__ dst,
                         const float* __restrict__ eattr, const float* __restrict__ dinv,
                         const float* __restrict__ hin, float* __restrict__ agg, int E)
{
    long long t = (long long)blockIdx.x * blockDim.x + threadIdx.x;
    long long e = t >> 5;
    int seg = (int)(t & 31);
    if (e >= E) return;
    int s = src[e], d = dst[e];
    float nrm = dinv[s] * eattr[2 * e + 1] * dinv[d];
    const float4 hv = *(const float4*)(hin + (size_t)s * DD + seg * 4);
    float* op = agg + (size_t)d * DD + seg * 4;
    atomicAdd(op + 0, nrm * hv.x);
    atomicAdd(op + 1, nrm * hv.y);
    atomicAdd(op + 2, nrm * hv.z);
    atomicAdd(op + 3, nrm * hv.w);
}

// ---------------------------------------------------------------- self loop + relu + BN stats
__global__ void selfloop_relu_stats(float* __restrict__ agg, const float* __restrict__ h,
                                    const float* __restrict__ dinv, const float* __restrict__ bias,
                                    float* __restrict__ stats, int n, int rows)
{
    int col = threadIdx.x;
    int r0 = blockIdx.x * rows;
    int r1 = min(r0 + rows, n);
    float b = bias[col];
    float s = 0.0f, s2 = 0.0f;
    for (int r = r0; r < r1; ++r) {
        float di = dinv[r];
        size_t idx = (size_t)r * DD + col;
        float v = agg[idx] + di * di * h[idx] + b;
        v = fmaxf(v, 0.0f);
        agg[idx] = v;
        s += v; s2 += v * v;
    }
    atomicAdd(&stats[col], s);
    atomicAdd(&stats[DD + col], s2);
}

__global__ void bn_apply(float* __restrict__ h, const float* __restrict__ stats,
                         const float* __restrict__ gamma, const float* __restrict__ beta,
                         long long n)    // n = rows
{
    long long i = (long long)blockIdx.x * blockDim.x + threadIdx.x;
    if (i >= n * DD) return;
    int col = (int)(i & (DD - 1));
    float invn = 1.0f / (float)n;
    float m = stats[col] * invn;
    float var = stats[DD + col] * invn - m * m;
    h[i] = (h[i] - m) * __frsqrt_rn(var + 1e-5f) * gamma[col] + beta[col];
}

// ---------------------------------------------------------------- pooling
__global__ void pool_counts(const int* __restrict__ batch, float* __restrict__ counts, int n) {
    int i = blockIdx.x * blockDim.x + threadIdx.x;
    if (i < n) atomicAdd(&counts[batch[i]], 1.0f);
}

__global__ void pool_acc(const float* __restrict__ h, const int* __restrict__ batch,
                         float* __restrict__ reps, int n) {
    long long t = (long long)blockIdx.x * blockDim.x + threadIdx.x;
    if (t >= (long long)n * DD) return;
    int row = (int)(t >> 7);
    atomicAdd(&reps[(size_t)batch[row] * DD + (int)(t & (DD - 1))], h[t]);
}

__global__ void pool_div(float* __restrict__ reps, const float* __restrict__ counts, int g) {
    int t = blockIdx.x * blockDim.x + threadIdx.x;
    if (t >= g * DD) return;
    reps[t] /= fmaxf(counts[t >> 7], 1.0f);
}

// ---------------------------------------------------------------- small dense (classifier tail)
__global__ void fc_small(const float* __restrict__ in, const float* __restrict__ W,
                         const float* __restrict__ bias, float* __restrict__ out,
                         int g, int K, int ncol, int do_relu)
{
    int t = blockIdx.x * blockDim.x + threadIdx.x;
    if (t >= g * ncol) return;
    int gi = t / ncol, j = t - gi * ncol;
    float acc = bias[j];
    for (int k = 0; k < K; ++k) acc += in[gi * K + k] * W[k * ncol + j];
    out[t] = do_relu ? fmaxf(acc, 0.0f) : acc;
}

// ================================================================ launch
extern "C" void kernel_launch(void* const* d_in, const int* in_sizes, int n_in,
                              void* d_out, int out_size, void* d_ws, size_t ws_size,
                              hipStream_t stream)
{
    const float* x     = (const float*)d_in[0];
    const int*   eidx  = (const int*)  d_in[1];
    const float* eattr = (const float*)d_in[2];
    const int*   batch = (const int*)  d_in[3];
    const float* Wg0 = (const float*)d_in[4];  const float* bg0 = (const float*)d_in[5];
    const float* g0  = (const float*)d_in[6];  const float* b0  = (const float*)d_in[7];
    const float* Wg1 = (const float*)d_in[8];  const float* bg1 = (const float*)d_in[9];
    const float* g1  = (const float*)d_in[10]; const float* b1  = (const float*)d_in[11];
    const float* Wl1 = (const float*)d_in[12]; const float* bl1 = (const float*)d_in[13];
    const float* Wl2 = (const float*)d_in[14]; const float* bl2 = (const float*)d_in[15];
    const float* Wc1 = (const float*)d_in[16]; const float* bc1 = (const float*)d_in[17];
    const float* Wc2 = (const float*)d_in[18]; const float* bc2 = (const float*)d_in[19];

    const int N = in_sizes[0] / DD;
    const int E = in_sizes[1] / 2;
    const int C = in_sizes[19];
    const int G = (out_size - N * DD) / (DD + C);

    const int* src = eidx;
    const int* dst = eidx + E;

    // ---- workspace carve-up
    char* ws = (char*)d_ws;
    size_t off = 0;
    _Float16* h16 = (_Float16*)(ws + off); off += (size_t)N * DD * sizeof(_Float16);
    float* bufA   = (float*)(ws + off);    off += (size_t)N * DD * sizeof(float);
    float* bufB   = (float*)(ws + off);    off += (size_t)N * DD * sizeof(float);
    float* deg    = (float*)(ws + off);    off += (size_t)N * sizeof(float);
    float* dinv   = (float*)(ws + off);    off += (size_t)N * sizeof(float);
    _Float16* wh0 = (_Float16*)(ws + off); off += DD * DD * sizeof(_Float16);
    _Float16* wh1 = (_Float16*)(ws + off); off += DD * DD * sizeof(_Float16);
    _Float16* wh2 = (_Float16*)(ws + off); off += DD * DD * sizeof(_Float16);
    _Float16* wh3 = (_Float16*)(ws + off); off += DD * DD * sizeof(_Float16);
    _Float16* wh4 = (_Float16*)(ws + off); off += DD * DD * sizeof(_Float16);
    float* stats  = (float*)(ws + off);    off += 2 * DD * sizeof(float);
    float* counts = (float*)(ws + off);    off += (size_t)G * sizeof(float);
    (void)ws_size; (void)n_in;

    float* out_h      = (float*)d_out;
    float* out_reps   = out_h + (size_t)N * DD;
    float* out_logits = out_reps + (size_t)G * DD;

    const long long ND = (long long)N * DD;
    const int TPB = 256;
    auto blk = [&](long long n) { return (unsigned)((n + TPB - 1) / TPB); };
    const int mblocks = (N + MT * 16 - 1) / (MT * 16);
    const int gblocks = (G + MT * 16 - 1) / (MT * 16);
    const int rowsPB  = 256;
    const int statBlk = (N + rowsPB - 1) / rowsPB;

    // ---- weights -> f16, transposed (col-major, K-contiguous per column)
    cvt_f16_t<<<blk(DD * DD), TPB, 0, stream>>>(Wg0, wh0, DD, DD);
    cvt_f16_t<<<blk(DD * DD), TPB, 0, stream>>>(Wg1, wh1, DD, DD);
    cvt_f16_t<<<blk(DD * DD), TPB, 0, stream>>>(Wl1, wh2, DD, DD);
    cvt_f16_t<<<blk(DD * DD), TPB, 0, stream>>>(Wl2, wh3, DD, DD);
    cvt_f16_t<<<blk(DD * DD), TPB, 0, stream>>>(Wc1, wh4, DD, DD);

    // ---- degree + symmetric norm
    fill_f32<<<blk(N), TPB, 0, stream>>>(deg, 1.0f, N);
    deg_acc<<<blk(E), TPB, 0, stream>>>(dst, eattr, deg, E);
    rsqrt_k<<<blk(N), TPB, 0, stream>>>(deg, dinv, N);

    // ================= conv layer 0 =================
    cvt_f16<<<blk(ND), TPB, 0, stream>>>(x, h16, ND);
    gemm_wmma_128<0, 0><<<mblocks, TPB, 0, stream>>>(h16, wh0, bufA, nullptr, N);
    zero_f32<<<blk(ND), TPB, 0, stream>>>(bufB, ND);
    edge_agg<<<blk((long long)E * 32), TPB, 0, stream>>>(src, dst, eattr, dinv, bufA, bufB, E);
    zero_f32<<<1, TPB, 0, stream>>>(stats, 2 * DD);
    selfloop_relu_stats<<<statBlk, DD, 0, stream>>>(bufB, bufA, dinv, bg0, stats, N, rowsPB);
    bn_apply<<<blk(ND), TPB, 0, stream>>>(bufB, stats, g0, b0, N);

    // ================= conv layer 1 =================
    cvt_f16<<<blk(ND), TPB, 0, stream>>>(bufB, h16, ND);
    gemm_wmma_128<0, 0><<<mblocks, TPB, 0, stream>>>(h16, wh1, bufA, nullptr, N);
    zero_f32<<<blk(ND), TPB, 0, stream>>>(bufB, ND);
    edge_agg<<<blk((long long)E * 32), TPB, 0, stream>>>(src, dst, eattr, dinv, bufA, bufB, E);
    zero_f32<<<1, TPB, 0, stream>>>(stats, 2 * DD);
    selfloop_relu_stats<<<statBlk, DD, 0, stream>>>(bufB, bufA, dinv, bg1, stats, N, rowsPB);
    bn_apply<<<blk(ND), TPB, 0, stream>>>(bufB, stats, g1, b1, N);

    // ================= node MLP =================
    cvt_f16<<<blk(ND), TPB, 0, stream>>>(bufB, h16, ND);
    gemm_wmma_128<1, 1><<<mblocks, TPB, 0, stream>>>(h16, wh2, bufA, bl1, N);   // relu(h@Wl1+bl1)
    cvt_f16<<<blk(ND), TPB, 0, stream>>>(bufA, h16, ND);
    gemm_wmma_128<0, 1><<<mblocks, TPB, 0, stream>>>(h16, wh3, out_h, bl2, N);  // h@Wl2+bl2 -> out h

    // ================= graph pooling =================
    zero_f32<<<blk((long long)G * DD), TPB, 0, stream>>>(out_reps, (long long)G * DD);
    zero_f32<<<1, TPB, 0, stream>>>(counts, G);
    pool_counts<<<blk(N), TPB, 0, stream>>>(batch, counts, N);
    pool_acc<<<blk(ND), TPB, 0, stream>>>(out_h, batch, out_reps, N);
    pool_div<<<blk((long long)G * DD), TPB, 0, stream>>>(out_reps, counts, G);

    // ================= classifier =================
    cvt_f16<<<blk((long long)G * DD), TPB, 0, stream>>>(out_reps, h16, (long long)G * DD);
    gemm_wmma_128<1, 1><<<gblocks, TPB, 0, stream>>>(h16, wh4, bufA, bc1, G);   // relu(reps@Wc1+bc1)
    fc_small<<<blk((long long)G * C), TPB, 0, stream>>>(bufA, Wc2, bc2, out_logits, G, DD, C, 0);
}
